// LSTM_keras_61151744360775
// MI455X (gfx1250) — compile-verified, matrix operand order checked by reading, not run on hardware
//
#include <hip/hip_runtime.h>
#include <hip/hip_bf16.h>
#include <stdint.h>

// ---------------------------------------------------------------------------
// Problem constants (match reference)
// ---------------------------------------------------------------------------
#define BB   256
#define TT   128
#define DD   128
#define HH   512
#define NCC  10
#define FOURH (4 * HH)

typedef unsigned short u16;
typedef unsigned int   u32;

typedef __attribute__((ext_vector_type(16))) __bf16 v16bf;
typedef __attribute__((ext_vector_type(8)))  float  v8f;

union FragBF { uint4 u[2]; v16bf v; };

// ---------------------------------------------------------------------------
// Helpers
// ---------------------------------------------------------------------------
__device__ __forceinline__ u16 f2bf(float f) {
    u32 u = __float_as_uint(f);
    u32 r = u + 0x7FFFu + ((u >> 16) & 1u);   // round-to-nearest-even
    return (u16)(r >> 16);
}
__device__ __forceinline__ float bf2f(u16 h) {
    return __uint_as_float(((u32)h) << 16);
}
__device__ __forceinline__ v8f wmma_bf16(v16bf a, v16bf b, v8f c) {
    return __builtin_amdgcn_wmma_f32_16x16x32_bf16(false, a, false, b, (short)0, c,
                                                   false, false);
}
// B fragment: packed contiguous 32 bytes per lane
__device__ __forceinline__ v16bf load_b_frag(const u16* bp) {
    FragBF f;
    const uint4* p = (const uint4*)bp;
    f.u[0] = p[0];
    f.u[1] = p[1];
    return f.v;
}
// A fragment from an LDS-staged 16xK bf16 panel (row-major, K halves/row).
// 16-bit A layout: lane<16 -> M=lane, K = {8g..8g+7, 16+8g..23+8g}, g=lane>>4
template <int KROW>
__device__ __forceinline__ v16bf load_a_frag(const u16* smA, int kb, unsigned lane) {
    unsigned ln = lane & 15u, g = lane >> 4;
    FragBF f;
    f.u[0] = *(const uint4*)(smA + ln * KROW + kb * 32 + 8 * g);
    f.u[1] = *(const uint4*)(smA + ln * KROW + kb * 32 + 16 + 8 * g);
    return f.v;
}

// ---------------------------------------------------------------------------
// Elementwise utility kernels
// ---------------------------------------------------------------------------
__global__ void f32_to_bf16_kernel(const float* __restrict__ src,
                                   u16* __restrict__ dst, int n) {
    int i = blockIdx.x * blockDim.x + threadIdx.x;
    if (i < n) dst[i] = f2bf(src[i]);
}
__global__ void zero_f32_kernel(float* __restrict__ p, int n) {
    int i = blockIdx.x * blockDim.x + threadIdx.x;
    if (i < n) p[i] = 0.0f;
}
__global__ void zero_u16_kernel(u16* __restrict__ p, int n) {
    int i = blockIdx.x * blockDim.x + threadIdx.x;
    if (i < n) p[i] = 0;
}

// Pack a row-major fp32 [K,N] weight matrix into bf16 WMMA B-fragment order:
// dst[((nblk*(K/32)+kblk)*32 + lane)*16 + j] = src[Kmap][nblk*16 + (lane&15)]
__global__ void pack_b_kernel(const float* __restrict__ src,
                              u16* __restrict__ dst, int K, int N) {
    int tid = blockIdx.x * blockDim.x + threadIdx.x;
    if (tid >= K * N) return;
    int j    = tid & 15;
    int lane = (tid >> 4) & 31;
    int tile = tid >> 9;
    int kT   = K >> 5;
    int kblk = tile % kT;
    int nblk = tile / kT;
    int g    = lane >> 4;
    int ln   = lane & 15;
    int kk   = kblk * 32 + ((j < 8 ? j : j + 8) + 8 * g);
    int nn   = nblk * 16 + ln;
    dst[tid] = f2bf(src[(size_t)kk * N + nn]);
}

// ---------------------------------------------------------------------------
// Big GEMM: C[M,N] = A[M,K](bf16) x Bpacked[K,N](bf16) + bias[N]   (fp32 out)
// Block = 256 thr (8 waves). Block tile: 16 rows x 512 cols.
// Wave tile: 16 x 64 (4 accumulators) -> one A-frag feeds 4 WMMAs.
// Full 16xK A panel staged in LDS ONCE; inner loop is barrier-free.
// ---------------------------------------------------------------------------
template <int K>
__global__ __launch_bounds__(256) void gemm_xw_kernel(
    const u16* __restrict__ A, const u16* __restrict__ Bp,
    const float* __restrict__ bias, float* __restrict__ C, int N) {
    __shared__ __align__(16) u16 smA[16 * K];
    const int tid       = threadIdx.x;
    const unsigned lane = tid & 31u;
    const unsigned wave = tid >> 5;
    const int mbase  = blockIdx.x * 16;
    const int nbase0 = blockIdx.y * 512 + wave * 64;
    const int nblk0  = nbase0 >> 4;
    constexpr int kT = K / 32;
    // stage full A panel (16 rows x K) once
    constexpr int chunks = 16 * K / 8;               // uint4 (8-half) chunks
    for (int idx = tid; idx < chunks; idx += 256) {
        int row = idx / (K / 8);
        int off = (idx % (K / 8)) * 8;
        *(uint4*)(smA + row * K + off) =
            *(const uint4*)(A + (size_t)(mbase + row) * K + off);
    }
    __syncthreads();
    v8f acc[4] = {{}, {}, {}, {}};
    for (int kb = 0; kb < kT; ++kb) {
        v16bf af = load_a_frag<K>(smA, kb, lane);
#pragma unroll
        for (int nt = 0; nt < 4; ++nt) {
            const u16* bp =
                Bp + ((size_t)(nblk0 + nt) * kT + kb) * 512 + lane * 16;
            acc[nt] = wmma_bf16(af, load_b_frag(bp), acc[nt]);
        }
    }
    const int ln = lane & 15;
    const int g  = lane >> 4;
#pragma unroll
    for (int nt = 0; nt < 4; ++nt) {
        int col  = nbase0 + nt * 16 + ln;
        float bv = bias[col];
#pragma unroll
        for (int r = 0; r < 8; ++r) {
            int m = mbase + r + 8 * g;
            C[(size_t)m * N + col] = acc[nt][r] + bv;
        }
    }
}

// ---------------------------------------------------------------------------
// One LSTM timestep:  z = xw_t + h_{t-1} @ U  (4 gates), then cell update.
// Block = 128 thr (4 waves). Block tile: 16 batch rows x 64 h cols.
// Full 16x512 h-panel staged in LDS once; inner loop: 16 k-steps x 4 WMMA.
// ---------------------------------------------------------------------------
__global__ __launch_bounds__(128) void lstm_step_kernel(
    const u16* __restrict__ hprev, int hprev_stride,  // bf16, elements
    const u16* __restrict__ Up,                        // packed U [H,4H]
    const float* __restrict__ xw,                      // + t*4H, row stride T*4H
    float* __restrict__ cbuf,                          // [B,H]
    u16* __restrict__ hout) {                          // + t*H, row stride T*H
    __shared__ __align__(16) u16 smA[16 * HH];         // 16 KB
    const int tid       = threadIdx.x;
    const unsigned lane = tid & 31u;
    const unsigned wave = tid >> 5;
    const int mbase = blockIdx.x * 16;                 // batch rows
    const int hcol  = blockIdx.y * 64 + wave * 16;     // h columns
    const int hblk  = hcol >> 4;
    constexpr int kT = HH / 32;                        // 16
    // stage full h_{t-1} panel (16 x 512) once
    for (int idx = tid; idx < 16 * HH / 8; idx += 128) {
        int row = idx >> 6;            // / (512/8)
        int off = (idx & 63) * 8;
        *(uint4*)(smA + row * HH + off) =
            *(const uint4*)(hprev + (size_t)(mbase + row) * hprev_stride + off);
    }
    __syncthreads();
    v8f acc[4] = {{}, {}, {}, {}};
    for (int kb = 0; kb < kT; ++kb) {
        v16bf af = load_a_frag<HH>(smA, kb, lane);
#pragma unroll
        for (int gate = 0; gate < 4; ++gate) {
            int nblk = gate * (HH / 16) + hblk;
            const u16* bp = Up + ((size_t)nblk * kT + kb) * 512 + lane * 16;
            acc[gate] = wmma_bf16(af, load_b_frag(bp), acc[gate]);
        }
    }
    const int ln  = lane & 15;
    const int g   = lane >> 4;
    const int col = hcol + ln;
#pragma unroll
    for (int r = 0; r < 8; ++r) {
        int b = mbase + r + 8 * g;
        const float* xr = xw + (size_t)b * (TT * FOURH);
        float zi = acc[0][r] + xr[0 * HH + col];
        float zf = acc[1][r] + xr[1 * HH + col];
        float zg = acc[2][r] + xr[2 * HH + col];
        float zo = acc[3][r] + xr[3 * HH + col];
        float iv = 1.0f / (1.0f + __expf(-zi));
        float fv = 1.0f / (1.0f + __expf(-zf));
        float ov = 1.0f / (1.0f + __expf(-zo));
        float gv = zg > 0.0f ? zg : 0.0f;               // cell activation = relu
        size_t ci = (size_t)b * HH + col;
        float cn = fv * cbuf[ci] + iv * gv;
        cbuf[ci] = cn;
        float hv = ov * (cn > 0.0f ? cn : 0.0f);        // output activation = relu
        hout[(size_t)b * (TT * HH) + col] = f2bf(hv);
    }
}

// ---------------------------------------------------------------------------
// Tiny dense head (scalar; negligible FLOPs)
// ---------------------------------------------------------------------------
__global__ void dense_bf16_kernel(const u16* __restrict__ A, int rowStride,
                                  const float* __restrict__ W,
                                  const float* __restrict__ bias,
                                  float* __restrict__ out, int Kd, int Nd,
                                  int doRelu) {
    int tid = blockIdx.x * blockDim.x + threadIdx.x;
    if (tid >= BB * Nd) return;
    int b = tid / Nd, n = tid % Nd;
    const u16* a = A + (size_t)b * rowStride;
    float s = bias[n];
    for (int k = 0; k < Kd; ++k) s += bf2f(a[k]) * W[(size_t)k * Nd + n];
    out[tid] = (doRelu && s < 0.0f) ? 0.0f : s;
}
__global__ void dense_f32_kernel(const float* __restrict__ A, int rowStride,
                                 const float* __restrict__ W,
                                 const float* __restrict__ bias,
                                 float* __restrict__ out, int Kd, int Nd,
                                 int doRelu) {
    int tid = blockIdx.x * blockDim.x + threadIdx.x;
    if (tid >= BB * Nd) return;
    int b = tid / Nd, n = tid % Nd;
    const float* a = A + (size_t)b * rowStride;
    float s = bias[n];
    for (int k = 0; k < Kd; ++k) s += a[k] * W[(size_t)k * Nd + n];
    out[tid] = (doRelu && s < 0.0f) ? 0.0f : s;
}
__global__ void softmax_kernel(const float* __restrict__ logits,
                               float* __restrict__ out, int rows, int C) {
    int b = blockIdx.x * blockDim.x + threadIdx.x;
    if (b >= rows) return;
    const float* l = logits + (size_t)b * C;
    float m = l[0];
    for (int i = 1; i < C; ++i) m = l[i] > m ? l[i] : m;
    float s = 0.0f;
    for (int i = 0; i < C; ++i) s += __expf(l[i] - m);
    float inv = 1.0f / s;
    for (int i = 0; i < C; ++i) out[(size_t)b * C + i] = __expf(l[i] - m) * inv;
}

// ---------------------------------------------------------------------------
// Host orchestration
// ---------------------------------------------------------------------------
extern "C" void kernel_launch(void* const* d_in, const int* in_sizes, int n_in,
                              void* d_out, int out_size, void* d_ws, size_t ws_size,
                              hipStream_t stream) {
    (void)in_sizes; (void)n_in; (void)out_size; (void)ws_size;
    const float* x   = (const float*)d_in[0];
    const float* W1  = (const float*)d_in[1];
    const float* U1  = (const float*)d_in[2];
    const float* b1  = (const float*)d_in[3];
    const float* W2  = (const float*)d_in[4];
    const float* U2  = (const float*)d_in[5];
    const float* b2  = (const float*)d_in[6];
    const float* W3  = (const float*)d_in[7];
    const float* U3  = (const float*)d_in[8];
    const float* b3  = (const float*)d_in[9];
    const float* Wf1 = (const float*)d_in[10];
    const float* bf1 = (const float*)d_in[11];
    const float* Wf2 = (const float*)d_in[12];
    const float* bf2 = (const float*)d_in[13];
    const float* Wf3 = (const float*)d_in[14];
    const float* bf3 = (const float*)d_in[15];

    char* w = (char*)d_ws;
    auto alloc = [&](size_t bytes) -> void* {
        void* p = (void*)w;
        w += (bytes + 255) & ~(size_t)255;
        return p;
    };
    u16*   x_bf   = (u16*)alloc((size_t)BB * TT * DD * 2);
    u16*   W1p    = (u16*)alloc((size_t)DD * FOURH * 2);
    u16*   W2p    = (u16*)alloc((size_t)HH * FOURH * 2);
    u16*   W3p    = (u16*)alloc((size_t)HH * FOURH * 2);
    u16*   U1p    = (u16*)alloc((size_t)HH * FOURH * 2);
    u16*   U2p    = (u16*)alloc((size_t)HH * FOURH * 2);
    u16*   U3p    = (u16*)alloc((size_t)HH * FOURH * 2);
    float* XW     = (float*)alloc((size_t)BB * TT * FOURH * 4);   // 256 MiB
    u16*   hseq1  = (u16*)alloc((size_t)BB * TT * HH * 2);        // 32 MiB
    u16*   hseq2  = (u16*)alloc((size_t)BB * TT * HH * 2);        // 32 MiB
    float* cbuf   = (float*)alloc((size_t)BB * HH * 4);
    u16*   zbuf   = (u16*)alloc((size_t)HH * 2);
    float* d1     = (float*)alloc((size_t)BB * 128 * 4);
    float* d2     = (float*)alloc((size_t)BB * 32 * 4);
    float* logits = (float*)alloc((size_t)BB * NCC * 4);
    u16*   hseq3  = hseq1;  // layer-1 output dead after layer-2 xW GEMM; reuse

    // 1. convert x to bf16
    {
        int n = BB * TT * DD;
        f32_to_bf16_kernel<<<(n + 255) / 256, 256, 0, stream>>>(x, x_bf, n);
    }
    // 2. pack all weight matrices into B-fragment order
    {
        int n = DD * FOURH;
        pack_b_kernel<<<(n + 255) / 256, 256, 0, stream>>>(W1, W1p, DD, FOURH);
        n = HH * FOURH;
        pack_b_kernel<<<(n + 255) / 256, 256, 0, stream>>>(U1, U1p, HH, FOURH);
        pack_b_kernel<<<(n + 255) / 256, 256, 0, stream>>>(W2, W2p, HH, FOURH);
        pack_b_kernel<<<(n + 255) / 256, 256, 0, stream>>>(U2, U2p, HH, FOURH);
        pack_b_kernel<<<(n + 255) / 256, 256, 0, stream>>>(W3, W3p, HH, FOURH);
        pack_b_kernel<<<(n + 255) / 256, 256, 0, stream>>>(U3, U3p, HH, FOURH);
        zero_u16_kernel<<<(HH + 255) / 256, 256, 0, stream>>>(zbuf, HH);
    }

    const u16*   Ain[3] = {x_bf, hseq1, hseq2};
    const u16*   Wp[3]  = {W1p, W2p, W3p};
    const u16*   Up[3]  = {U1p, U2p, U3p};
    const float* bl[3]  = {b1, b2, b3};
    u16*         hs[3]  = {hseq1, hseq2, hseq3};

    for (int l = 0; l < 3; ++l) {
        // xW + b for all timesteps at once: [B*T, K] @ [K, 4H]
        dim3 g((BB * TT) / 16, FOURH / 512);
        if (l == 0)
            gemm_xw_kernel<DD><<<g, 256, 0, stream>>>(Ain[l], Wp[l], bl[l], XW,
                                                      FOURH);
        else
            gemm_xw_kernel<HH><<<g, 256, 0, stream>>>(Ain[l], Wp[l], bl[l], XW,
                                                      FOURH);
        zero_f32_kernel<<<(BB * HH + 255) / 256, 256, 0, stream>>>(cbuf, BB * HH);
        for (int t = 0; t < TT; ++t) {
            const u16* hp = (t == 0) ? zbuf : (hs[l] + (size_t)(t - 1) * HH);
            int hstride   = (t == 0) ? 0 : TT * HH;
            dim3 gs(BB / 16, HH / 64);
            lstm_step_kernel<<<gs, 128, 0, stream>>>(hp, hstride, Up[l],
                                                     XW + (size_t)t * FOURH, cbuf,
                                                     hs[l] + (size_t)t * HH);
        }
    }

    // dense head + softmax
    {
        int n = BB * 128;
        dense_bf16_kernel<<<(n + 255) / 256, 256, 0, stream>>>(
            hseq3 + (size_t)(TT - 1) * HH, TT * HH, Wf1, bf1, d1, HH, 128, 1);
        n = BB * 32;
        dense_f32_kernel<<<(n + 255) / 256, 256, 0, stream>>>(d1, 128, Wf2, bf2, d2,
                                                              128, 32, 1);
        n = BB * NCC;
        dense_f32_kernel<<<(n + 255) / 256, 256, 0, stream>>>(d2, 32, Wf3, bf3,
                                                              logits, 32, NCC, 0);
        softmax_kernel<<<(BB + 255) / 256, 256, 0, stream>>>(logits, (float*)d_out,
                                                             BB, NCC);
    }
}
